// RoFormerSelfAttention_66829691126482
// MI455X (gfx1250) — matrix operand order repaired and checked by reading, not compiled, not run on hardware
//
#include <hip/hip_runtime.h>
#include <hip/hip_bf16.h>

// ---------------------------------------------------------------------------
// RoFormer self-attention for MI455X (gfx1250, wave32, WMMA bf16 + async LDS)
// B=2, S=2048, D=1024, H=16, HD=64
// ---------------------------------------------------------------------------

typedef __attribute__((ext_vector_type(16))) __bf16 v16bf;
typedef __attribute__((ext_vector_type(8)))  __bf16 v8bf;
typedef __attribute__((ext_vector_type(8)))  float  v8f;

#define BDIM 1024
#define SEQ  2048
#define NB   2
#define NH   16
#define HDIM 64
#define ROWS (NB*SEQ)                 // 4096
#define MATELTS ((size_t)ROWS*BDIM)   // 4194304 elements per Q/K/V matrix

static __device__ __forceinline__ __bf16 f32_bf16(float f) {
    unsigned u = __builtin_bit_cast(unsigned, f);
    unsigned r = u + 0x7FFFu + ((u >> 16) & 1u);   // round-to-nearest-even
    unsigned short h = (unsigned short)(r >> 16);
    return __builtin_bit_cast(__bf16, h);
}

static __device__ __forceinline__ v16bf cat8(v8bf lo, v8bf hi) {
    return __builtin_shufflevector(lo, hi, 0,1,2,3,4,5,6,7,8,9,10,11,12,13,14,15);
}

static __device__ __forceinline__ v8f wmma_bf16(v16bf a, v16bf b, v8f c) {
    return __builtin_amdgcn_wmma_f32_16x16x32_bf16(false, a, false, b, (short)0, c,
                                                   false, false);
}

// gfx1250 async global->LDS 16B copy (per-lane gather), tracked by ASYNCcnt.
static __device__ __forceinline__ void async_copy16(void* lds_ptr, const void* gptr) {
    unsigned lds_off = (unsigned)(uintptr_t)lds_ptr;  // flat LDS addr: low 32 bits
    asm volatile("global_load_async_to_lds_b128 %0, %1, off"
                 :: "v"(lds_off), "v"(gptr) : "memory");
}
static __device__ __forceinline__ void wait_async0() {
    asm volatile("s_wait_asynccnt 0x0" ::: "memory");
}

// --------------------------- f32 -> bf16 copy ------------------------------
__global__ void cvt_x(const float* __restrict__ x, __bf16* __restrict__ xb, int n) {
    int i = blockIdx.x * 256 + threadIdx.x;
    if (i < n) xb[i] = f32_bf16(x[i]);
}

// ------------------- weights: f32 -> bf16, transposed ----------------------
// wt[z][n][k] = w_z[k][n]  so the WMMA B operand reads contiguous K per lane.
__global__ void cvt_wt(const float* __restrict__ w0, const float* __restrict__ w1,
                       const float* __restrict__ w2, __bf16* __restrict__ wt) {
    int z = blockIdx.y;
    const float* w = (z == 0) ? w0 : (z == 1) ? w1 : w2;
    int i = blockIdx.x * 256 + threadIdx.x;   // n fastest -> coalesced read
    int k = i >> 10, n = i & 1023;
    wt[(size_t)z * BDIM * BDIM + (size_t)n * BDIM + k] = f32_bf16(w[i]);
}

// --------------------- fused QKV GEMM + bias + RoPE ------------------------
// grid: (32 Mblocks, 16 Nblocks, 3 matrices), block: 256 = 8 waves.
// B tiles (shared by all 8 waves) staged in LDS via async copies, double-buffered.
#define WSTRIDE 72   // padded LDS row stride (bf16): 144B, 16B-aligned, conflict-free

__global__ __launch_bounds__(256)
void qkv_gemm(const __bf16* __restrict__ xb, const __bf16* __restrict__ wt,
              const float* __restrict__ bq, const float* __restrict__ bk,
              const float* __restrict__ bvv, __bf16* __restrict__ qkvb) {
    __shared__ alignas(32) __bf16 WbS[2][64 * WSTRIDE];  // 64 cols x 64 k, 2 buffers

    const int z = blockIdx.z;
    const __bf16* w   = wt + (size_t)z * BDIM * BDIM;
    const float* bias = (z == 0) ? bq : (z == 1) ? bk : bvv;
    __bf16* dst = qkvb + (size_t)z * MATELTS;

    const int lane = threadIdx.x & 31;
    const int wv   = threadIdx.x >> 5;
    const int hf   = lane >> 4;
    const int ln   = lane & 15;
    const int m0   = blockIdx.x * 128 + wv * 16;
    const int n0   = blockIdx.y * 64;

    const int tn  = threadIdx.x >> 3;          // 0..31 (col group for stage copy)
    const int tc8 = (threadIdx.x & 7) * 8;     // 8-elem column offset

    // Prologue: stage 0 (8KB = 512 x b128 block-wide, 2 per thread).
#pragma unroll
    for (int r = 0; r < 2; ++r) {
        int n = tn + r * 32;
        async_copy16(&WbS[0][n * WSTRIDE + tc8], w + (size_t)(n0 + n) * BDIM + tc8);
    }

    v8f acc[4] = {};
    const __bf16* arow = xb + (size_t)(m0 + ln) * BDIM;   // A layout: row = lane%16
    int p = 0;

    for (int kb = 0; kb < BDIM; kb += 64) {
        wait_async0();            // stage p resident (this wave's share)
        __syncthreads();          // all waves' shares resident; prev reads of p^1 done
        if (kb + 64 < BDIM) {     // prefetch next stage into p^1 (overlaps WMMAs)
#pragma unroll
            for (int r = 0; r < 2; ++r) {
                int n = tn + r * 32;
                async_copy16(&WbS[p ^ 1][n * WSTRIDE + tc8],
                             w + (size_t)(n0 + n) * BDIM + kb + 64 + tc8);
            }
        }
#pragma unroll
        for (int ks = 0; ks < 64; ks += 32) {
            // A 16x32: elements 0..15 -> k = hf*16+e
            v16bf a = cat8(*(const v8bf*)(arow + kb + ks + hf * 16),
                           *(const v8bf*)(arow + kb + ks + hf * 16 + 8));
#pragma unroll
            for (int nt = 0; nt < 4; ++nt) {
                // B 32x16 from LDS: lane col n = ln, elements k = hf*16 + e
                const __bf16* bp = &WbS[p][(nt * 16 + ln) * WSTRIDE + ks + hf * 16];
                acc[nt] = wmma_bf16(a, *(const v16bf*)bp, acc[nt]);
            }
        }
        p ^= 1;
    }

    // Epilogue: bias, interleaved RoPE (pairs live in adjacent lanes), store bf16.
#pragma unroll
    for (int nt = 0; nt < 4; ++nt) {
        int gn = n0 + nt * 16 + ln;
        float bsv = bias[gn];
        int h = gn >> 6, hd = gn & 63;
        float inv = __powf(10000.0f, -(float)(hd & ~1) * (1.0f / 64.0f));
#pragma unroll
        for (int v = 0; v < 8; ++v) {
            int gm = m0 + v + 8 * hf;          // C layout: m = v + 8*half
            int bb = gm >> 11, s = gm & 2047;
            float t = acc[nt][v] + bsv;
            float partner = __shfl_xor(t, 1, 32);
            float r = t;
            if (z < 2) {                       // RoPE on Q and K only (uniform branch)
                float ang = (float)s * inv;
                float sn, cs;
                __sincosf(ang, &sn, &cs);
                r = (hd & 1) ? (t * cs + partner * sn) : (t * cs - partner * sn);
            }
            dst[(((size_t)(bb * NH + h) * SEQ + s) * HDIM) + hd] = f32_bf16(r);
        }
    }
}

// ------------------------- flash attention ---------------------------------
// grid: (S/128 = 16, B*H = 32), block 256 = 8 waves. Wave owns 16 query rows.
// K chunk staged via async copy (shared by all waves); V chunk manually
// transposed into LDS; P staged per-wave through LDS for C->A layout change.
#define VSTRIDE 40
#define PSTRIDE 40
#define KSTRIDE 72

__global__ __launch_bounds__(256)
void attn(const __bf16* __restrict__ Qg, const __bf16* __restrict__ Kg,
          const __bf16* __restrict__ Vg, float* __restrict__ out) {
    __shared__ alignas(32) __bf16 VtS[2][HDIM * VSTRIDE];  // V^T chunk (d major)
    __shared__ alignas(32) __bf16 KbS[2][32 * KSTRIDE];    // K chunk (row major)
    __shared__ alignas(32) __bf16 PbS[8][16 * PSTRIDE];    // per-wave P staging

    const int lane = threadIdx.x & 31;
    const int wv   = threadIdx.x >> 5;
    const int hf   = lane >> 4;
    const int ln   = lane & 15;
    const int bh   = blockIdx.y;               // b*16 + h
    const int q0   = blockIdx.x * 128 + wv * 16;

    const __bf16* Q = Qg + (size_t)bh * SEQ * HDIM;
    const __bf16* K = Kg + (size_t)bh * SEQ * HDIM;
    const __bf16* V = Vg + (size_t)bh * SEQ * HDIM;

    const int tkv = threadIdx.x >> 3;          // 0..31 (K row for stage copy)
    const int tc8 = (threadIdx.x & 7) * 8;

    // Prologue: async K chunk 0 -> KbS[0] (4KB = 256 x b128 block-wide).
    async_copy16(&KbS[0][tkv * KSTRIDE + tc8], K + (size_t)tkv * HDIM + tc8);

    // Q tile in A layout, K-dim = d (0..63) split in two WMMA k-steps.
    const __bf16* qrow = Q + (size_t)(q0 + ln) * HDIM;
    v16bf a0 = cat8(*(const v8bf*)(qrow + hf * 16),      *(const v8bf*)(qrow + hf * 16 + 8));
    v16bf a1 = cat8(*(const v8bf*)(qrow + 32 + hf * 16), *(const v8bf*)(qrow + 40 + hf * 16));

    v8f acc[4] = {};
    float mrow[8], lrow[8];
#pragma unroll
    for (int v = 0; v < 8; ++v) { mrow[v] = -1e30f; lrow[v] = 0.0f; }

    const float scale = 0.125f;                // 1/sqrt(64)
    __bf16* pw = &PbS[wv][0];
    int pb = 0;

    for (int j = 0; j < SEQ; j += 32) {
        // Cooperative transposed V chunk store: Vt[d][kv], coalesced global reads.
#pragma unroll
        for (int i = 0; i < 8; ++i) {
            int e = threadIdx.x + i * 256;
            int kv = e >> 6, d = e & 63;
            VtS[pb][d * VSTRIDE + kv] = V[(size_t)(j + kv) * HDIM + d];
        }
        wait_async0();            // K chunk j resident (own share)
        __syncthreads();          // all shares resident; prev reads of pb^1 done

        if (j + 32 < SEQ) {       // prefetch next K chunk (safe: after barrier)
            async_copy16(&KbS[pb ^ 1][tkv * KSTRIDE + tc8],
                         K + (size_t)(j + 32 + tkv) * HDIM + tc8);
            __builtin_prefetch(V + (size_t)(j + 32 + tkv) * HDIM + tc8, 0, 3);
        }

        // Scores: S = Q K^T from LDS K (B layout: lane col n = ln, k contiguous).
        const __bf16* k0 = &KbS[pb][ln * KSTRIDE + hf * 16];
        const __bf16* k1 = &KbS[pb][(16 + ln) * KSTRIDE + hf * 16];
        v8f s0 = {}, s1 = {};
        s0 = wmma_bf16(a0, *(const v16bf*)(k0),      s0);
        s0 = wmma_bf16(a1, *(const v16bf*)(k0 + 32), s0);
        s1 = wmma_bf16(a0, *(const v16bf*)(k1),      s1);
        s1 = wmma_bf16(a1, *(const v16bf*)(k1 + 32), s1);

        // Online softmax (row reductions across 16 lanes of each half-wave).
#pragma unroll
        for (int v = 0; v < 8; ++v) {
            float x0 = s0[v] * scale, x1 = s1[v] * scale;
            float cm = fmaxf(x0, x1);
            cm = fmaxf(cm, __shfl_xor(cm, 1, 32));
            cm = fmaxf(cm, __shfl_xor(cm, 2, 32));
            cm = fmaxf(cm, __shfl_xor(cm, 4, 32));
            cm = fmaxf(cm, __shfl_xor(cm, 8, 32));
            float mn = fmaxf(mrow[v], cm);
            float alpha = __expf(mrow[v] - mn);
            float p0 = __expf(x0 - mn), p1 = __expf(x1 - mn);
            float rs = p0 + p1;
            rs += __shfl_xor(rs, 1, 32);
            rs += __shfl_xor(rs, 2, 32);
            rs += __shfl_xor(rs, 4, 32);
            rs += __shfl_xor(rs, 8, 32);
            lrow[v] = lrow[v] * alpha + rs;
            mrow[v] = mn;
#pragma unroll
            for (int nt = 0; nt < 4; ++nt) acc[nt][v] *= alpha;
            int m = v + 8 * hf;                 // C-layout row -> stage P in LDS
            pw[m * PSTRIDE + ln]      = f32_bf16(p0);
            pw[m * PSTRIDE + 16 + ln] = f32_bf16(p1);
        }

        // Re-read P in A layout (same-wave LDS, in order).
        const __bf16* pr = pw + ln * PSTRIDE;
        v16bf pa = cat8(*(const v8bf*)(pr + hf * 16), *(const v8bf*)(pr + hf * 16 + 8));

        // ctx += P * V : B from transposed-V LDS (n = d column, k = kv contiguous).
#pragma unroll
        for (int nt = 0; nt < 4; ++nt) {
            const __bf16* vp = &VtS[pb][(nt * 16 + ln) * VSTRIDE + hf * 16];
            acc[nt] = wmma_bf16(pa, *(const v16bf*)vp, acc[nt]);
        }
        pb ^= 1;
    }

    // Normalize and scatter to [B,S,D] fp32 output.
    const int b = bh >> 4, h = bh & 15;
#pragma unroll
    for (int nt = 0; nt < 4; ++nt) {
#pragma unroll
        for (int v = 0; v < 8; ++v) {
            int s = q0 + v + 8 * hf;
            int d = nt * 16 + ln;
            out[((size_t)(b * SEQ + s)) * BDIM + h * HDIM + d] = acc[nt][v] / lrow[v];
        }
    }
}

// ---------------------------------------------------------------------------
extern "C" void kernel_launch(void* const* d_in, const int* in_sizes, int n_in,
                              void* d_out, int out_size, void* d_ws, size_t ws_size,
                              hipStream_t stream) {
    const float* x  = (const float*)d_in[0];
    const float* Wq = (const float*)d_in[1];
    const float* bq = (const float*)d_in[2];
    const float* Wk = (const float*)d_in[3];
    const float* bk = (const float*)d_in[4];
    const float* Wv = (const float*)d_in[5];
    const float* bv = (const float*)d_in[6];
    float* out = (float*)d_out;

    // Workspace (bf16 elems): X (4M) | Wt x3 (3M) | QKV x3 (12M) ~= 39.8 MB
    __bf16* xb   = (__bf16*)d_ws;
    __bf16* wt   = xb + MATELTS;
    __bf16* qkvb = wt + 3 * (size_t)BDIM * BDIM;

    cvt_x<<<(int)(MATELTS / 256), 256, 0, stream>>>(x, xb, (int)MATELTS);
    cvt_wt<<<dim3(BDIM * BDIM / 256, 3), 256, 0, stream>>>(Wq, Wk, Wv, wt);
    qkv_gemm<<<dim3(ROWS / 128, BDIM / 64, 3), 256, 0, stream>>>(xb, wt, bq, bk, bv, qkvb);
    attn<<<dim3(SEQ / 128, NB * NH), 256, 0, stream>>>(qkvb, qkvb + MATELTS,
                                                       qkvb + 2 * MATELTS, out);
}
